// SS4RecLayer_76931454206423
// MI455X (gfx1250) — matrix operand order repaired and checked by reading, not compiled, not run on hardware
//
#include <hip/hip_runtime.h>
#include <math.h>

typedef __attribute__((ext_vector_type(16))) _Float16 v16h;
typedef __attribute__((ext_vector_type(8)))  float    v8f;

#define DM    128
#define DI    256
#define PSTATE 16
#define LSEQ  2048
#define BSZ   32
#define NBL   (BSZ * LSEQ)         // 65536 rows
#define DTRK  8
#define LN_EPS 1e-5f
#define S5_NCH 16                  // chunks for S5 scan
#define S5_LC  (LSEQ / S5_NCH)     // 128

// ---------------- WMMA fragment helpers (gfx1250 wave32, 16x16x32 f16) ----------

// A fragment: 16x32 (MxK), f32 source row-major, convert to f16.
// lane<16: M=lane, elements i -> K = (i&7) + (i>>3)*16 ; lanes>=16 add +8.
__device__ __forceinline__ v16h load_a_f16(const float* __restrict__ A, int lda,
                                           int m0, int k0, int lane) {
  const int m  = m0 + (lane & 15);
  const int hs = (lane >> 4) & 1;
  v16h a;
#pragma unroll
  for (int i = 0; i < 16; ++i) {
    const int k = k0 + (i & 7) + ((i >> 3) << 4) + (hs << 3);
    a[i] = (_Float16)A[(size_t)m * lda + k];
  }
  return a;
}

// B fragment: 32x16 (KxN), f32 source row-major [K x N].
// lane<16: N=lane, element i -> K = (i&7) + (i>>3)*16 ; lanes>=16 add +8.
__device__ __forceinline__ v16h load_b_f16(const float* __restrict__ B, int ldb,
                                           int k0, int n0, int lane) {
  const int n  = n0 + (lane & 15);
  const int hs = (lane >> 4) & 1;
  v16h b;
#pragma unroll
  for (int i = 0; i < 16; ++i) {
    const int k = k0 + (i & 7) + ((i >> 3) << 4) + (hs << 3);
    b[i] = (_Float16)B[(size_t)k * ldb + n];
  }
  return b;
}

// Guarded B loader for ragged N (x_proj has only 40 cols).
__device__ __forceinline__ v16h load_b_f16_guard(const float* __restrict__ B, int ldb,
                                                 int ncols, int k0, int n0, int lane) {
  const int n  = n0 + (lane & 15);
  const int hs = (lane >> 4) & 1;
  v16h b;
#pragma unroll
  for (int i = 0; i < 16; ++i) {
    const int k = k0 + (i & 7) + ((i >> 3) << 4) + (hs << 3);
    float v = (n < ncols) ? B[(size_t)k * ldb + n] : 0.0f;
    b[i] = (_Float16)v;
  }
  return b;
}

__device__ __forceinline__ float wave_row_sum(float s) {
#pragma unroll
  for (int off = 16; off >= 1; off >>= 1) s += __shfl_xor(s, off, 32);
  return s;
}

// Row LayerNorm over a 16x128 LDS tile; wave w normalizes rows 2w, 2w+1.
__device__ __forceinline__ void tile_ln_store(float (*tile)[DM + 4],
                                              const float* __restrict__ g,
                                              const float* __restrict__ be,
                                              float* __restrict__ out,
                                              int m0, int wave, int lane) {
#pragma unroll
  for (int rr = 2 * wave; rr <= 2 * wave + 1; ++rr) {
    float s = 0.f;
#pragma unroll
    for (int j = 0; j < 4; ++j) s += tile[rr][lane + 32 * j];
    s = wave_row_sum(s);
    const float mu = s * (1.0f / DM);
    float v = 0.f;
#pragma unroll
    for (int j = 0; j < 4; ++j) {
      const float d = tile[rr][lane + 32 * j] - mu;
      v += d * d;
    }
    v = wave_row_sum(v);
    const float rs = rsqrtf(v * (1.0f / DM) + LN_EPS);
#pragma unroll
    for (int j = 0; j < 4; ++j) {
      const int ci = lane + 32 * j;
      out[(size_t)(m0 + rr) * DM + ci] = (tile[rr][ci] - mu) * rs * g[ci] + be[ci];
    }
  }
}

// ---------------- K0: derived parameters -----------------------------------------
// prm layout (floats): Ad[16]@0  Astar[16]@16  AdLc[16]@32  Bd[16*128]@64  Aneg[256*16]@2176
__global__ void k_prep(const float* __restrict__ s5_A_log,
                       const float* __restrict__ s5_log_dt,
                       const float* __restrict__ s5_B,
                       const float* __restrict__ m_A_log,
                       float* __restrict__ prm) {
  const int t = threadIdx.x;
  if (t < PSTATE) {
    const float A  = -expf(s5_A_log[t]);
    const float dt = expf(s5_log_dt[t]);
    prm[t]        = expf(A * dt);                 // Ad
    prm[16 + t]   = A * dt;                       // Astar
    prm[32 + t]   = expf(A * dt * (float)S5_LC);  // Ad^Lc
  }
  for (int idx = t; idx < PSTATE * DM; idx += blockDim.x) {
    const int p = idx >> 7;
    const float A  = -expf(s5_A_log[p]);
    const float dt = expf(s5_log_dt[p]);
    const float Ad = expf(A * dt);
    prm[64 + idx] = ((Ad - 1.0f) / A) * s5_B[idx];          // Bd[p][h]
  }
  for (int idx = t; idx < DI * PSTATE; idx += blockDim.x)
    prm[2176 + idx] = -expf(m_A_log[idx]);                  // Aneg[d][p]
}

// ---------------- K1: Bu = x @ Bd^T  (65536x128 * 128x16) -----------------------
__global__ void k_s5_bu(const float* __restrict__ x, const float* __restrict__ prm,
                        float* __restrict__ bu) {
  const int gid = blockIdx.x * blockDim.x + threadIdx.x;
  const int bl = gid >> 4, p = gid & 15;
  const float* __restrict__ Bd = prm + 64 + p * DM;
  const float* __restrict__ xr = x + (size_t)bl * DM;
  float acc = 0.f;
#pragma unroll 8
  for (int h = 0; h < DM; ++h) acc = fmaf(xr[h], Bd[h], acc);
  bu[(size_t)bl * PSTATE + p] = acc;
}

// ---------------- K2: S5 local chunk scan (in place) ----------------------------
__global__ void k_s5_scan_local(float* __restrict__ bu, const float* __restrict__ prm,
                                float* __restrict__ carry) {
  const int gid = blockIdx.x * blockDim.x + threadIdx.x;   // 8192
  const int p = gid & 15, chunk = (gid >> 4) & (S5_NCH - 1), b = gid >> 8;
  const float Ad = prm[p];
  size_t base = ((size_t)(b * LSEQ + chunk * S5_LC)) * PSTATE + p;
  float xv = 0.f;
  for (int j = 0; j < S5_LC; ++j) {
    xv = fmaf(Ad, xv, bu[base]);
    bu[base] = xv;
    base += PSTATE;
  }
  carry[(size_t)(b * S5_NCH + chunk) * PSTATE + p] = xv;
}

// ---------------- K3: chunk-carry exclusive prefix ------------------------------
__global__ void k_s5_carry(const float* __restrict__ carry, const float* __restrict__ prm,
                           float* __restrict__ pref) {
  const int t = threadIdx.x;        // 512 = 32 b * 16 p
  const int b = t >> 4, p = t & 15;
  const float AdLc = prm[32 + p];
  float pf = 0.f;
  for (int c = 0; c < S5_NCH; ++c) {
    const size_t idx = (size_t)(b * S5_NCH + c) * PSTATE + p;
    const float cc = carry[idx];
    pref[idx] = pf;
    pf = fmaf(AdLc, pf, cc);
  }
}

// ---------------- K4: apply carry fix: xs += pref * Ad^(lloc+1) ------------------
__global__ void k_s5_fix(float* __restrict__ bu, const float* __restrict__ prm,
                         const float* __restrict__ pref) {
  const int gid = blockIdx.x * blockDim.x + threadIdx.x;
  const int bl = gid >> 4, p = gid & 15;
  const int b = bl >> 11, l = bl & (LSEQ - 1);
  const int chunk = l >> 7, lloc = l & (S5_LC - 1);
  const float pf = pref[(size_t)(b * S5_NCH + chunk) * PSTATE + p];
  const float Astar = prm[16 + p];
  bu[(size_t)bl * PSTATE + p] += pf * __expf(Astar * (float)(lloc + 1));
}

// ---------------- K5: y = xs@C^T + x*D, residual, LN1 ---------------------------
__global__ void k_s5_out_ln1(const float* __restrict__ bu, const float* __restrict__ x,
                             const float* __restrict__ C,   // [128,16]
                             const float* __restrict__ D,
                             const float* __restrict__ g1, const float* __restrict__ b1,
                             float* __restrict__ x1) {
  __shared__ float xsh[16 * PSTATE];
  __shared__ float tile[16][DM + 4];
  const int t = threadIdx.x, m0 = blockIdx.x * 16;
  xsh[t] = bu[(size_t)m0 * PSTATE + t];
  __syncthreads();
#pragma unroll
  for (int e = 0; e < 8; ++e) {
    const int idx = t + 256 * e;
    const int r = idx >> 7, col = idx & (DM - 1);
    const float* Crow = C + col * PSTATE;
    float acc = 0.f;
#pragma unroll
    for (int p = 0; p < PSTATE; ++p) acc = fmaf(xsh[r * PSTATE + p], Crow[p], acc);
    const float xv = x[(size_t)(m0 + r) * DM + col];
    tile[r][col] = acc + xv * D[col] + xv;      // s5_out + residual
  }
  __syncthreads();
  tile_ln_store(tile, g1, b1, x1, m0, t >> 5, t & 31);
}

// ---------------- K6: in_proj GEMM (WMMA f16) -----------------------------------
__global__ void k_in_proj(const float* __restrict__ x1, const float* __restrict__ W,
                          float* __restrict__ xi, float* __restrict__ z) {
  const int wave = threadIdx.x >> 5, lane = threadIdx.x & 31;
  const int m0 = blockIdx.x * 16;
  const int n0 = (blockIdx.y * 8 + wave) * 16;      // 0..511
  v8f c = {};
#pragma unroll
  for (int k0 = 0; k0 < DM; k0 += 32) {
    v16h a = load_a_f16(x1, DM, m0, k0, lane);
    v16h b = load_b_f16(W, 2 * DI, k0, n0, lane);
    c = __builtin_amdgcn_wmma_f32_16x16x32_f16(false, a, false, b, (short)0, c, false, false);
  }
  const int col = n0 + (lane & 15);
  const int rbase = m0 + (((lane >> 4) & 1) << 3);
  float* dst = (col < DI) ? xi : z;
  const int cc = (col < DI) ? col : col - DI;
#pragma unroll
  for (int r = 0; r < 8; ++r) dst[(size_t)(rbase + r) * DI + cc] = c[r];
}

// ---------------- K7: causal depthwise conv (w=4) + bias + SiLU -----------------
__global__ void k_conv_silu(const float* __restrict__ xi, const float* __restrict__ w,
                            const float* __restrict__ cb, float* __restrict__ xc) {
  const int d = threadIdx.x;
  const int b = blockIdx.x >> 11, l = blockIdx.x & (LSEQ - 1);
  float acc = cb[d];
#pragma unroll
  for (int j = 0; j < 4; ++j) {
    const int lj = l - 3 + j;
    if (lj >= 0) acc = fmaf(w[d * 4 + j], xi[((size_t)(b * LSEQ + lj)) * DI + d], acc);
  }
  const float s = acc / (1.0f + __expf(-acc));      // silu
  xc[((size_t)(b * LSEQ + l)) * DI + d] = s;
}

// ---------------- K8: x_proj GEMM (WMMA f16, N=40 ragged, stride 48) ------------
__global__ void k_x_proj(const float* __restrict__ xc, const float* __restrict__ W,
                         float* __restrict__ dbc) {
  const int wave = threadIdx.x >> 5, lane = threadIdx.x & 31;   // 3 waves
  const int m0 = blockIdx.x * 16;
  const int n0 = wave * 16;
  v8f c = {};
#pragma unroll
  for (int k0 = 0; k0 < DI; k0 += 32) {
    v16h a = load_a_f16(xc, DI, m0, k0, lane);
    v16h b = load_b_f16_guard(W, DTRK + 2 * PSTATE, DTRK + 2 * PSTATE, k0, n0, lane);
    c = __builtin_amdgcn_wmma_f32_16x16x32_f16(false, a, false, b, (short)0, c, false, false);
  }
  const int col = n0 + (lane & 15);
  const int rbase = m0 + (((lane >> 4) & 1) << 3);
  if (col < DTRK + 2 * PSTATE) {
#pragma unroll
    for (int r = 0; r < 8; ++r) dbc[(size_t)(rbase + r) * 48 + col] = c[r];
  }
}

// ---------------- K9: dt = softplus(dbc[:, :8] @ dt_w + dt_b) -------------------
__global__ void k_dt(const float* __restrict__ dbc, const float* __restrict__ dtw,
                     const float* __restrict__ dtb, float* __restrict__ dt) {
  __shared__ float r8[DTRK];
  const int d = threadIdx.x;
  const size_t bl = blockIdx.x;
  if (d < DTRK) r8[d] = dbc[bl * 48 + d];
  __syncthreads();
  float v = dtb[d];
#pragma unroll
  for (int r = 0; r < DTRK; ++r) v = fmaf(r8[r], dtw[r * DI + d], v);
  const float sp = (v > 20.0f) ? v : log1pf(__expf(v));
  dt[bl * DI + d] = sp;
}

// ---------------- K10: Mamba selective scan, (b,d,p)-parallel --------------------
__global__ void k_scan(const float* __restrict__ u, const float* __restrict__ dt,
                       const float* __restrict__ dbc, const float* __restrict__ prm,
                       const float* __restrict__ Dp, float* __restrict__ y) {
  const int b = blockIdx.x >> 4, dg = blockIdx.x & 15;
  const int p = threadIdx.x & 15;
  const int d = dg * 16 + (threadIdx.x >> 4);
  const float A  = prm[2176 + d * PSTATE + p];
  const float Dd = Dp[d];
  const float* __restrict__ ub  = u   + (size_t)b * LSEQ * DI + d;
  const float* __restrict__ db  = dt  + (size_t)b * LSEQ * DI + d;
  const float* __restrict__ Bb  = dbc + (size_t)b * LSEQ * 48 + DTRK + p;
  const float* __restrict__ Cb  = dbc + (size_t)b * LSEQ * 48 + DTRK + PSTATE + p;
  float* __restrict__ yb = y + (size_t)b * LSEQ * DI + d;

  float h = 0.f;
  float uc = ub[0], dc = db[0], Bc = Bb[0], Cc = Cb[0];
  for (int t = 0; t < LSEQ; ++t) {
    float un = 0.f, dn = 0.f, Bn = 0.f, Cn = 0.f;
    if (t + 1 < LSEQ) {                       // prefetch next step into registers
      const size_t o = (size_t)(t + 1) * DI, o2 = (size_t)(t + 1) * 48;
      un = ub[o]; dn = db[o]; Bn = Bb[o2]; Cn = Cb[o2];
    }
    if (t + 8 < LSEQ) {                       // deep prefetch -> global_prefetch_b8
      __builtin_prefetch(ub + (size_t)(t + 8) * DI, 0, 1);
      __builtin_prefetch(db + (size_t)(t + 8) * DI, 0, 1);
      __builtin_prefetch(Bb + (size_t)(t + 8) * 48, 0, 1);
    }
    const float dA = __expf(dc * A);
    h = fmaf(dA, h, dc * uc * Bc);
    float yp = h * Cc;
    yp += __shfl_xor(yp, 1, 32);
    yp += __shfl_xor(yp, 2, 32);
    yp += __shfl_xor(yp, 4, 32);
    yp += __shfl_xor(yp, 8, 32);              // sum over 16 p-lanes
    if (p == 0) yb[(size_t)t * DI] = fmaf(uc, Dd, yp);
    uc = un; dc = dn; Bc = Bn; Cc = Cn;
  }
}

// ---------------- K11: gated out_proj GEMM (WMMA) + residual + LN2 --------------
__global__ void k_out_ln2(const float* __restrict__ y, const float* __restrict__ z,
                          const float* __restrict__ W,   // [256,128]
                          const float* __restrict__ x1,
                          const float* __restrict__ g2, const float* __restrict__ b2,
                          float* __restrict__ out) {
  __shared__ float tile[16][DM + 4];
  const int wave = threadIdx.x >> 5, lane = threadIdx.x & 31;
  const int m0 = blockIdx.x * 16;
  const int n0 = wave * 16;
  const int m  = m0 + (lane & 15);
  const int hs = (lane >> 4) & 1;
  v8f c = {};
#pragma unroll
  for (int k0 = 0; k0 < DI; k0 += 32) {
    v16h a;
#pragma unroll
    for (int i = 0; i < 16; ++i) {            // fused SiLU gate on A fragment
      const int k = k0 + (i & 7) + ((i >> 3) << 4) + (hs << 3);
      const float yv = y[(size_t)m * DI + k];
      const float zv = z[(size_t)m * DI + k];
      a[i] = (_Float16)(yv * (zv / (1.0f + __expf(-zv))));
    }
    v16h b = load_b_f16(W, DM, k0, n0, lane);
    c = __builtin_amdgcn_wmma_f32_16x16x32_f16(false, a, false, b, (short)0, c, false, false);
  }
  const int col = n0 + (lane & 15);
  const int rbase = hs << 3;
#pragma unroll
  for (int r = 0; r < 8; ++r)
    tile[rbase + r][col] = c[r] + x1[(size_t)(m0 + rbase + r) * DM + col];
  __syncthreads();
  tile_ln_store(tile, g2, b2, out, m0, wave, lane);
}

// ---------------- launch ---------------------------------------------------------
extern "C" void kernel_launch(void* const* d_in, const int* in_sizes, int n_in,
                              void* d_out, int out_size, void* d_ws, size_t ws_size,
                              hipStream_t stream) {
  const float* x        = (const float*)d_in[0];
  const float* s5_A_log = (const float*)d_in[1];
  const float* s5_B     = (const float*)d_in[2];
  const float* s5_C     = (const float*)d_in[3];
  const float* s5_D     = (const float*)d_in[4];
  const float* s5_logdt = (const float*)d_in[5];
  const float* ln1_g    = (const float*)d_in[6];
  const float* ln1_b    = (const float*)d_in[7];
  const float* ln2_g    = (const float*)d_in[8];
  const float* ln2_b    = (const float*)d_in[9];
  const float* in_proj  = (const float*)d_in[10];
  const float* conv_w   = (const float*)d_in[11];
  const float* conv_b   = (const float*)d_in[12];
  const float* x_proj   = (const float*)d_in[13];
  const float* dt_w     = (const float*)d_in[14];
  const float* dt_b     = (const float*)d_in[15];
  const float* m_A_log  = (const float*)d_in[16];
  const float* m_D      = (const float*)d_in[17];
  const float* out_proj = (const float*)d_in[18];
  float* out = (float*)d_out;

  char* ws = (char*)d_ws;
  size_t o = 0;
  auto al = [&](size_t bytes) { size_t r = o; o += (bytes + 255) & ~(size_t)255; return r; };
  float* prm   = (float*)(ws + al(64 * 1024));
  float* bu    = (float*)(ws + al((size_t)NBL * PSTATE * 4));
  float* carry = (float*)(ws + al(8192 * 4));
  float* pref  = (float*)(ws + al(8192 * 4));
  float* x1    = (float*)(ws + al((size_t)NBL * DM * 4));
  float* bufA  = (float*)(ws + al((size_t)NBL * DI * 4));  // xi, later y_scan
  float* bufB  = (float*)(ws + al((size_t)NBL * DI * 4));  // z
  float* bufC  = (float*)(ws + al((size_t)NBL * DI * 4));  // conv+silu output
  float* dtbuf = (float*)(ws + al((size_t)NBL * DI * 4));
  float* dbc   = (float*)(ws + al((size_t)NBL * 48 * 4));

  k_prep<<<1, 256, 0, stream>>>(s5_A_log, s5_logdt, s5_B, m_A_log, prm);
  k_s5_bu<<<NBL * PSTATE / 256, 256, 0, stream>>>(x, prm, bu);
  k_s5_scan_local<<<BSZ * S5_NCH * PSTATE / 256, 256, 0, stream>>>(bu, prm, carry);
  k_s5_carry<<<1, 512, 0, stream>>>(carry, prm, pref);
  k_s5_fix<<<NBL * PSTATE / 256, 256, 0, stream>>>(bu, prm, pref);
  k_s5_out_ln1<<<NBL / 16, 256, 0, stream>>>(bu, x, s5_C, s5_D, ln1_g, ln1_b, x1);
  k_in_proj<<<dim3(NBL / 16, 4), 256, 0, stream>>>(x1, in_proj, bufA, bufB);
  k_conv_silu<<<NBL, 256, 0, stream>>>(bufA, conv_w, conv_b, bufC);
  k_x_proj<<<NBL / 16, 96, 0, stream>>>(bufC, x_proj, dbc);
  k_dt<<<NBL, 256, 0, stream>>>(dbc, dt_w, dt_b, dtbuf);
  k_scan<<<BSZ * 16, 256, 0, stream>>>(bufC, dtbuf, dbc, prm, m_D, bufA);
  k_out_ln2<<<NBL / 16, 256, 0, stream>>>(bufA, bufB, out_proj, x1, ln2_g, ln2_b, out);
}
